// BiLSTM_53558242181231
// MI455X (gfx1250) — matrix operand order repaired and checked by reading, not compiled, not run on hardware
//
#include <hip/hip_runtime.h>
#include <hip/hip_bf16.h>
#include <math.h>

typedef __attribute__((ext_vector_type(16))) _Float16 v16h;
typedef __attribute__((ext_vector_type(8)))  _Float16 v8h;
typedef __attribute__((ext_vector_type(8)))  float    v8f;

#define B_TOT 1024
#define T_LEN 512
#define H1    64
#define H2    32

__device__ __forceinline__ void wait_ds0() {
    asm volatile("s_wait_dscnt 0x0" ::: "memory");
}

// Branch-free fast activations: v_exp_f32 + v_rcp_f32 only, no EXEC games.
__device__ __forceinline__ float sigm(float x) {
    return __builtin_amdgcn_rcpf(1.f + __expf(-x));
}
__device__ __forceinline__ float tanh_fast(float x) {
    float a = __builtin_fabsf(x);
    float e = __expf(-2.f * a);                    // e in (0, 1]
    float t = (1.f - e) * __builtin_amdgcn_rcpf(1.f + e);
    return __builtin_copysignf(t, x);
}

__device__ __forceinline__ v8f wmma16(v16h a, v16h b, v8f c) {
    // v_wmma_f32_16x16x32_f16: D = A(16x32) * B(32x16) + C
    return __builtin_amdgcn_wmma_f32_16x16x32_f16(false, a, false, b, (short)0, c, false, false);
}

// ---- Fragment loaders (layouts per CDNA5 ISA 7.12.2) ----
// A (16xK f16, M=lane&15): lanes<16 hold K = k0+[0..7] and k0+[16..23];
// lanes>=16 hold K = k0+[8..15] and k0+[24..31].  p is row-major [M][strideH] halves.
__device__ __forceinline__ v16h load_a(const _Float16* p, int strideH, int k0, int lane) {
    int m = lane & 15, hi = lane >> 4;
    const _Float16* q = p + m * strideH + k0 + hi * 8;
    v8h lo = *(const v8h*)q;
    v8h hv = *(const v8h*)(q + 16);
    return __builtin_shufflevector(lo, hv, 0,1,2,3,4,5,6,7,8,9,10,11,12,13,14,15);
}
// B (32x16 f16, N=lane&15): lanes<16 hold K = k0+[0..7] and k0+[16..23];
// lanes>=16 hold K = k0+[8..15] and k0+[24..31].  p is [N][strideH] halves (i.e. weight row-major [n][k]).
__device__ __forceinline__ v16h load_b(const _Float16* p, int strideH, int n0, int k0, int lane) {
    int n = n0 + (lane & 15), hi = lane >> 4;
    const _Float16* q = p + n * strideH + k0 + hi * 8;
    v8h lo = *(const v8h*)q;
    v8h hv = *(const v8h*)(q + 16);
    return __builtin_shufflevector(lo, hv, 0,1,2,3,4,5,6,7,8,9,10,11,12,13,14,15);
}

// =====================================================================
// Kernel 1: bidirectional layer-1 LSTM.  64 blocks x 64 threads
// (wave0 = forward, wave1 = reverse).  Writes h1[t][b][0:64]=fwd,
// [64:128]=rev as f16.
// =====================================================================
__global__ __launch_bounds__(64)
void lstm_layer1(const float* __restrict__ x,
                 const float* __restrict__ wih_f, const float* __restrict__ whh_f,
                 const float* __restrict__ b_f,
                 const float* __restrict__ wih_r, const float* __restrict__ whh_r,
                 const float* __restrict__ b_r,
                 _Float16* __restrict__ h1) {
    __shared__ __align__(128) float    xs[16 * T_LEN];        // 32 KB  x tile [row][t]
    __shared__ __align__(128) _Float16 wB[2][4 * H1 * H1];    // 64 KB  Whh f16 [dir][n][k]
    __shared__ __align__(128) _Float16 hst[2][16 * H1];       //  4 KB  h stage [dir][row][k]

    const int tid  = threadIdx.x;
    const int lane = tid & 31;
    const int wv   = tid >> 5;          // 0 = fwd, 1 = rev
    const int bt   = blockIdx.x;        // batch tile (16 rows)
    const int hi   = lane >> 4;

    for (int i = tid; i < 16 * T_LEN; i += 64) {
        int r = i >> 9, t = i & (T_LEN - 1);
        xs[i] = x[(size_t)(bt * 16 + r) * T_LEN + t];
    }
    {
        const float* w = wv ? whh_r : whh_f;
        _Float16* wd = &wB[wv][0];
        for (int i = lane; i < 4 * H1 * H1; i += 32) wd[i] = (_Float16)w[i];
        _Float16* hd = &hst[wv][0];
        for (int i = lane; i < 16 * H1; i += 32) hd[i] = (_Float16)0.f;
    }
    __syncthreads();

    const float* wih = wv ? wih_r : wih_f;
    const float* bb  = wv ? b_r   : b_f;
    float wih_v[16], bias_v[16];
#pragma unroll
    for (int n = 0; n < 16; n++) {
        int col = n * 16 + (lane & 15);
        wih_v[n]  = wih[col];           // Wih1 is [256,1]
        bias_v[n] = bb[col];
    }

    v8f c[4];
#pragma unroll
    for (int n = 0; n < 4; n++) { v8f z = {0.f,0.f,0.f,0.f,0.f,0.f,0.f,0.f}; c[n] = z; }

    const _Float16* wBd = &wB[wv][0];
    _Float16*       hsd = &hst[wv][0];

#pragma unroll 1
    for (int s = 0; s < T_LEN; s++) {
        const int t = wv ? (T_LEN - 1 - s) : s;
        float xv[8];
#pragma unroll
        for (int r = 0; r < 8; r++) xv[r] = xs[(r + hi * 8) * T_LEN + t];   // LDS broadcast
        v16h a0 = load_a(hsd, H1, 0,  lane);    // previous h (f16)
        v16h a1 = load_a(hsd, H1, 32, lane);

#pragma unroll
        for (int g = 0; g < 4; g++) {           // hidden tile group (16 cols each)
            v8f z[4];                           // i, f, g, o tiles
#pragma unroll
            for (int q = 0; q < 4; q++) {
                int nt = q * 4 + g;             // i:0-3 f:4-7 g:8-11 o:12-15
                v8f cc;
#pragma unroll
                for (int r = 0; r < 8; r++) cc[r] = fmaf(wih_v[nt], xv[r], bias_v[nt]);
                cc = wmma16(a0, load_b(wBd, H1, nt * 16, 0,  lane), cc);
                cc = wmma16(a1, load_b(wBd, H1, nt * 16, 32, lane), cc);
                z[q] = cc;
            }
            v8f hh;
#pragma unroll
            for (int r = 0; r < 8; r++) {
                float ci = sigm(z[1][r]) * c[g][r] + sigm(z[0][r]) * tanh_fast(z[2][r]);
                c[g][r] = ci;
                hh[r] = sigm(z[3][r]) * tanh_fast(ci);
            }
#pragma unroll
            for (int r = 0; r < 8; r++)
                hsd[(r + hi * 8) * H1 + g * 16 + (lane & 15)] = (_Float16)hh[r];
        }
        wait_ds0();
        // stream h tile to global h1 (coalesced 128b stores)
        {
            int row = lane >> 1, half = lane & 1;
            const v8h* src = (const v8h*)(hsd + row * H1 + half * 32);
            _Float16* dst = h1 + ((size_t)t * B_TOT + bt * 16 + row) * 128 + wv * 64 + half * 32;
            v8h d0 = src[0], d1 = src[1];
            *(v8h*)dst = d0;
            *(v8h*)(dst + 8) = d1;
        }
    }
}

// =====================================================================
// Kernel 2: layer-2 forward scan over h1, one reverse step at t=T-1,
// then the FC head.  64 blocks x 32 threads (1 wave per batch tile).
// =====================================================================
__global__ __launch_bounds__(32)
void lstm_layer2_head(const _Float16* __restrict__ h1,
                      const float* __restrict__ wih2f, const float* __restrict__ whh2f,
                      const float* __restrict__ b2f,
                      const float* __restrict__ wih2r, const float* __restrict__ b2r,
                      const float* __restrict__ wfc1, const float* __restrict__ bfc1,
                      const float* __restrict__ wout, const float* __restrict__ bout,
                      float* __restrict__ out) {
    __shared__ __align__(128) _Float16 w2f[128 * 160];   // 40 KB: k<128 Wih2f, k>=128 Whh2f
    __shared__ __align__(128) _Float16 w2r[128 * 128];   // 32 KB: Wih2r
    __shared__ __align__(128) _Float16 wfc[64 * 64];     //  8 KB
    __shared__ __align__(128) _Float16 h2s[16 * H2];     //  1 KB  h2 stage
    __shared__ __align__(128) _Float16 hcat[16 * 64];    //  2 KB  [h2f | h2r]
    __shared__ __align__(128) float    red[16 * 16];     //  1 KB

    const int lane = threadIdx.x & 31;
    const int bt   = blockIdx.x;
    const int hi   = lane >> 4;

    for (int i = lane; i < 128 * 128; i += 32) {
        int n = i >> 7, k = i & 127;
        w2f[n * 160 + k] = (_Float16)wih2f[i];
        w2r[i] = (_Float16)wih2r[i];
    }
    for (int i = lane; i < 128 * 32; i += 32) {
        int n = i >> 5, k = i & 31;
        w2f[n * 160 + 128 + k] = (_Float16)whh2f[i];
    }
    for (int i = lane; i < 64 * 64; i += 32) wfc[i] = (_Float16)wfc1[i];
    for (int i = lane; i < 16 * H2; i += 32) h2s[i] = (_Float16)0.f;
    wait_ds0();
    __syncthreads();

    float b2f_v[8], b2r_v[8];
#pragma unroll
    for (int n = 0; n < 8; n++) {
        int col = n * 16 + (lane & 15);
        b2f_v[n] = b2f[col];
        b2r_v[n] = b2r[col];
    }

    v8f c[2];
    { v8f z = {0.f,0.f,0.f,0.f,0.f,0.f,0.f,0.f}; c[0] = z; c[1] = z; }

#pragma unroll 1
    for (int t = 0; t < T_LEN; t++) {
        const _Float16* h1t = h1 + ((size_t)t * B_TOT + bt * 16) * 128;
        v16h aIn[4];
#pragma unroll
        for (int k = 0; k < 4; k++) aIn[k] = load_a(h1t, 128, k * 32, lane);  // global_load_b128
        v16h aR = load_a(h2s, H2, 0, lane);                                   // previous h2

#pragma unroll
        for (int g = 0; g < 2; g++) {           // hidden tiles (H2=32 -> 2 tiles)
            v8f z[4];
#pragma unroll
            for (int q = 0; q < 4; q++) {
                int nt = q * 2 + g;             // i:0-1 f:2-3 g:4-5 o:6-7
                v8f cc;
#pragma unroll
                for (int r = 0; r < 8; r++) cc[r] = b2f_v[nt];
#pragma unroll
                for (int k = 0; k < 4; k++)
                    cc = wmma16(aIn[k], load_b(w2f, 160, nt * 16, k * 32, lane), cc);
                cc = wmma16(aR, load_b(w2f, 160, nt * 16, 128, lane), cc);
                z[q] = cc;
            }
            v8f hh;
#pragma unroll
            for (int r = 0; r < 8; r++) {
                float ci = sigm(z[1][r]) * c[g][r] + sigm(z[0][r]) * tanh_fast(z[2][r]);
                c[g][r] = ci;
                hh[r] = sigm(z[3][r]) * tanh_fast(ci);
            }
#pragma unroll
            for (int r = 0; r < 8; r++)
                h2s[(r + hi * 8) * H2 + g * 16 + (lane & 15)] = (_Float16)hh[r];
        }
    }
    wait_ds0();

    // hcat[:,0:32] = h2f final
    for (int i = lane; i < 16 * H2; i += 32)
        hcat[(i >> 5) * 64 + (i & 31)] = h2s[i];

    // reverse layer-2, single step at t=T-1 with h0=c0=0 -> hcat[:,32:64]
    {
        const _Float16* h1t = h1 + ((size_t)(T_LEN - 1) * B_TOT + bt * 16) * 128;
        v16h aIn[4];
#pragma unroll
        for (int k = 0; k < 4; k++) aIn[k] = load_a(h1t, 128, k * 32, lane);
#pragma unroll
        for (int g = 0; g < 2; g++) {
            v8f z[4];
#pragma unroll
            for (int q = 0; q < 4; q++) {
                int nt = q * 2 + g;
                v8f cc;
#pragma unroll
                for (int r = 0; r < 8; r++) cc[r] = b2r_v[nt];
#pragma unroll
                for (int k = 0; k < 4; k++)
                    cc = wmma16(aIn[k], load_b(w2r, 128, nt * 16, k * 32, lane), cc);
                z[q] = cc;
            }
#pragma unroll
            for (int r = 0; r < 8; r++) {
                float ci = sigm(z[0][r]) * tanh_fast(z[2][r]);      // c0 = 0
                float hh = sigm(z[3][r]) * tanh_fast(ci);
                hcat[(r + hi * 8) * 64 + 32 + g * 16 + (lane & 15)] = (_Float16)hh;
            }
        }
    }
    wait_ds0();

    // head: relu(hcat @ Wfc1.T + bfc1) @ Wout.T + bout -> sigmoid
    {
        v16h a0 = load_a(hcat, 64, 0,  lane);
        v16h a1 = load_a(hcat, 64, 32, lane);
        float wout_v[4], bfc_v[4], part[8];
#pragma unroll
        for (int n = 0; n < 4; n++) {
            int col = n * 16 + (lane & 15);
            wout_v[n] = wout[col];
            bfc_v[n]  = bfc1[col];
        }
#pragma unroll
        for (int r = 0; r < 8; r++) part[r] = 0.f;
#pragma unroll
        for (int n = 0; n < 4; n++) {
            v8f cc;
#pragma unroll
            for (int r = 0; r < 8; r++) cc[r] = bfc_v[n];
            cc = wmma16(a0, load_b(wfc, 64, n * 16, 0,  lane), cc);
            cc = wmma16(a1, load_b(wfc, 64, n * 16, 32, lane), cc);
#pragma unroll
            for (int r = 0; r < 8; r++) part[r] += fmaxf(cc[r], 0.f) * wout_v[n];
        }
#pragma unroll
        for (int r = 0; r < 8; r++) red[(r + hi * 8) * 16 + (lane & 15)] = part[r];
        wait_ds0();
        if (lane < 16) {
            float s = bout[0];
#pragma unroll
            for (int j = 0; j < 16; j++) s += red[lane * 16 + j];
            out[bt * 16 + lane] = sigm(s);
        }
    }
}

extern "C" void kernel_launch(void* const* d_in, const int* in_sizes, int n_in,
                              void* d_out, int out_size, void* d_ws, size_t ws_size,
                              hipStream_t stream) {
    (void)in_sizes; (void)n_in; (void)out_size; (void)ws_size;
    const float* x     = (const float*)d_in[0];
    const float* wih1f = (const float*)d_in[1];
    const float* whh1f = (const float*)d_in[2];
    const float* b1f   = (const float*)d_in[3];
    const float* wih1r = (const float*)d_in[4];
    const float* whh1r = (const float*)d_in[5];
    const float* b1r   = (const float*)d_in[6];
    const float* wih2f = (const float*)d_in[7];
    const float* whh2f = (const float*)d_in[8];
    const float* b2f   = (const float*)d_in[9];
    const float* wih2r = (const float*)d_in[10];
    // d_in[11] (whh2r) unused: reverse layer-2 contributes only its first step (h0=0)
    const float* b2r   = (const float*)d_in[12];
    const float* wfc1  = (const float*)d_in[13];
    const float* bfc1  = (const float*)d_in[14];
    const float* wout  = (const float*)d_in[15];
    const float* bout  = (const float*)d_in[16];

    _Float16* h1 = (_Float16*)d_ws;   // needs T*B*128*2 = 128 MB scratch

    lstm_layer1<<<B_TOT / 16, 64, 0, stream>>>(x, wih1f, whh1f, b1f,
                                               wih1r, whh1r, b1r, h1);
    lstm_layer2_head<<<B_TOT / 16, 32, 0, stream>>>(h1, wih2f, whh2f, b2f,
                                                    wih2r, b2r, wfc1, bfc1,
                                                    wout, bout, (float*)d_out);
}
